// SPIntraAttModuleV2_33346126086740
// MI455X (gfx1250) — matrix operand order repaired and checked by reading, not compiled, not run on hardware
//
#include <hip/hip_runtime.h>
#include <hip/hip_bf16.h>
#include <math.h>

typedef __attribute__((ext_vector_type(16))) _Float16 v16h;
typedef __attribute__((ext_vector_type(8)))  float    v8f;

#define B_   2
#define C_   256
#define HW_  16384      // 128*128
#define S_   256
#define NH_  8
#define HD_  32         // head dim = 256/8
#define T_   128        // top-k
#define SCALE_ 0.17677669529663687f   // (256/8)^-0.5
#define EPS_ 1e-5f

// ---------------- WMMA fragment loaders (wave32, 16x16x32 f16) ----------------

// A: 16x32 f16, row-major source with leading dim ld (elements).
__device__ inline v16h load_A(const _Float16* __restrict__ A, int ld, int lane) {
  int hl = lane >> 4, m = lane & 15;
  const _Float16* row = A + m * ld;
  v16h a;
#pragma unroll
  for (int v = 0; v < 8; ++v) {
    int k = hl * 16 + 2 * v;
    a[2 * v]     = row[k];
    a[2 * v + 1] = row[k + 1];
  }
  return a;
}

// B: 32x16 f16, source B[k][n] row-major with leading dim ld.
__device__ inline v16h load_B(const _Float16* __restrict__ Bp, int ld, int lane) {
  int hl = lane >> 4, n = lane & 15;
  v16h b;
#pragma unroll
  for (int v = 0; v < 8; ++v) {
    int k = hl * 16 + 2 * v;
    b[2 * v]     = Bp[k * ld + n];
    b[2 * v + 1] = Bp[(k + 1) * ld + n];
  }
  return b;
}

// B: 32x16 f16 where B[k][n] = Bt[n*ld + k]  (Bt is 16 rows x 32 cols, row-major)
__device__ inline v16h load_B_T(const _Float16* __restrict__ Bt, int ld, int lane) {
  int hl = lane >> 4, n = lane & 15;
  const _Float16* row = Bt + n * ld;
  v16h b;
#pragma unroll
  for (int v = 0; v < 8; ++v) {
    int k = hl * 16 + 2 * v;
    b[2 * v]     = row[k];
    b[2 * v + 1] = row[k + 1];
  }
  return b;
}

// ---- gfx1250 async global->LDS copy (16B per lane), tracked on ASYNCcnt ----
__device__ inline void async_copy_b128(unsigned lds_off, const void* gptr) {
  unsigned long long ga = (unsigned long long)(uintptr_t)gptr;
  asm volatile("global_load_async_to_lds_b128 %0, %1, off"
               :: "v"(lds_off), "v"(ga) : "memory");
}
__device__ inline void wait_async0() {
  asm volatile("s_wait_asynccnt 0" ::: "memory");
}

// ---------------- Kernel 1: LayerNorm over C, emit f16 activations ----------------

__global__ void ln_kernel(const float* __restrict__ x,
                          const float* __restrict__ w,
                          const float* __restrict__ bia,
                          _Float16* __restrict__ xnh) {
  int wid = threadIdx.x >> 5;
  int lane = threadIdx.x & 31;
  int pos = blockIdx.x * 8 + wid;          // [0, B*HW)
  int b = pos / HW_, n = pos % HW_;
  const float* xp = x + (size_t)b * C_ * HW_ + n;
  float vals[8];
  float s = 0.f, ss = 0.f;
#pragma unroll
  for (int j = 0; j < 8; ++j) {
    float t = xp[(size_t)(lane + 32 * j) * HW_];
    vals[j] = t; s += t; ss += t * t;
  }
#pragma unroll
  for (int off = 16; off > 0; off >>= 1) {
    s  += __shfl_xor(s, off, 32);
    ss += __shfl_xor(ss, off, 32);
  }
  float mu  = s * (1.0f / C_);
  float var = ss * (1.0f / C_) - mu * mu;
  float rs  = rsqrtf(var + EPS_);
  _Float16* op = xnh + (size_t)b * C_ * HW_ + n;
#pragma unroll
  for (int j = 0; j < 8; ++j) {
    int ch = lane + 32 * j;
    float y = (vals[j] - mu) * rs * w[ch] + bia[ch];
    op[(size_t)ch * HW_] = (_Float16)y;
  }
}

// ---------------- Kernel 2: f32 -> f16 weight conversion ----------------

__global__ void cvt_f16_kernel(const float* __restrict__ src,
                               _Float16* __restrict__ dst, int n) {
  int i = blockIdx.x * blockDim.x + threadIdx.x;
  if (i < n) dst[i] = (_Float16)src[i];
}

// ---------------- Kernel 3: block-tiled projection GEMM ----------------
// out[b,o,n] = sum_c W[o,c] * X[b,c,n]; 128x128 output tile per block,
// K staged in 32-chunks through LDS using async global->LDS b128 copies.

template <int MODE>   // 0,1 -> f16 out (q,k); 2 -> f32 out (v)
__global__ void proj_kernel(const _Float16* __restrict__ Wh,
                            const _Float16* __restrict__ Xh,
                            _Float16* __restrict__ out_h,
                            float* __restrict__ out_f) {
  extern __shared__ char smem[];
  _Float16* Wt = (_Float16*)smem;            // [128][32] f16 = 8KB
  _Float16* Xt = Wt + 128 * 32;              // [32][128] f16 = 8KB
  unsigned lds_base = (unsigned)(uintptr_t)smem;   // LDS offset taken from the low 32 bits of the pointer

  int tid = threadIdx.x;
  int wid = tid >> 5, lane = tid & 31;
  int wr = wid >> 1, wc = wid & 1;           // 4x2 wave grid over 128x128 tile

  int blk = blockIdx.x;                      // b * (2*128) + ob*128 + nb
  int b  = blk / (2 * 128);
  int rem = blk % (2 * 128);
  int o_base = (rem / 128) * 128;
  int n_base = (rem % 128) * 128;
  const _Float16* Xb = Xh + (size_t)b * C_ * HW_;

  // staging indices (8 halves = 16B per thread per tile)
  int ew  = tid * 8;
  int wrow = ew >> 5, wcol = ew & 31;        // W tile: 128 rows x 32 cols
  int xrow = ew >> 7, xcol = ew & 127;       // X tile: 32 rows x 128 cols
  unsigned w_lds = lds_base + (unsigned)(ew * 2);
  unsigned x_lds = lds_base + 8192u + (unsigned)(ew * 2);

  v8f acc[2][4];
#pragma unroll
  for (int mi = 0; mi < 2; ++mi)
#pragma unroll
    for (int ni = 0; ni < 4; ++ni) acc[mi][ni] = (v8f){};

  for (int kc = 0; kc < C_; kc += 32) {
    __syncthreads();   // previous chunk fully consumed
    async_copy_b128(w_lds, Wh + (size_t)(o_base + wrow) * C_ + kc + wcol);
    async_copy_b128(x_lds, Xb + (size_t)(kc + xrow) * HW_ + n_base + xcol);
    wait_async0();
    __syncthreads();   // all async writes visible to all waves

    v16h a0 = load_A(Wt + (wr * 32 + 0)  * 32, 32, lane);
    v16h a1 = load_A(Wt + (wr * 32 + 16) * 32, 32, lane);
#pragma unroll
    for (int ni = 0; ni < 4; ++ni) {
      v16h bm = load_B(Xt + wc * 64 + ni * 16, 128, lane);
      acc[0][ni] = __builtin_amdgcn_wmma_f32_16x16x32_f16(false, a0, false, bm,
                                                          (short)0, acc[0][ni], false, false);
      acc[1][ni] = __builtin_amdgcn_wmma_f32_16x16x32_f16(false, a1, false, bm,
                                                          (short)0, acc[1][ni], false, false);
    }
  }

  int hl = lane >> 4, nn = lane & 15;
#pragma unroll
  for (int mi = 0; mi < 2; ++mi) {
#pragma unroll
    for (int ni = 0; ni < 4; ++ni) {
#pragma unroll
      for (int r = 0; r < 8; ++r) {
        int m = r + hl * 8;
        int o = o_base + wr * 32 + mi * 16 + m;
        int n = n_base + wc * 64 + ni * 16 + nn;
        size_t oi = (size_t)b * C_ * HW_ + (size_t)o * HW_ + n;
        if (MODE == 2) out_f[oi] = acc[mi][ni][r];
        else           out_h[oi] = (_Float16)acc[mi][ni][r];
      }
    }
  }
}

// ---------------- Kernel 4: top-128 per affinity row (iterative argmax) ----------------

__global__ void topk_kernel(const float* __restrict__ aff,
                            float* __restrict__ sims,
                            int* __restrict__ idxo) {
  extern __shared__ char smem[];
  float* row = (float*)smem;                          // HW_ floats
  float* rv  = (float*)(smem + (size_t)HW_ * 4);      // 256
  int*   ri  = (int*)  (smem + (size_t)HW_ * 4 + 1024);
  int tid = threadIdx.x;
  int r   = blockIdx.x;                               // b*S + s
  const float* src = aff + (size_t)r * HW_;
  for (int i = tid; i < HW_; i += 256) row[i] = src[i];
  __syncthreads();
  for (int t = 0; t < T_; ++t) {
    float best = -INFINITY; int bi = 0;
    for (int i = tid; i < HW_; i += 256) {
      float v = row[i];
      if (v > best) { best = v; bi = i; }
    }
    rv[tid] = best; ri[tid] = bi;
    __syncthreads();
    for (int s = 128; s > 0; s >>= 1) {
      if (tid < s) {
        float v2 = rv[tid + s];
        if (v2 > rv[tid] || (v2 == rv[tid] && ri[tid + s] < ri[tid])) {
          rv[tid] = v2; ri[tid] = ri[tid + s];
        }
      }
      __syncthreads();
    }
    if (tid == 0) {
      sims[(size_t)r * T_ + t] = rv[0];
      idxo[(size_t)r * T_ + t] = ri[0];
      row[ri[0]] = -INFINITY;
    }
    __syncthreads();
  }
}

// ---------------- Kernel 5: per-(b,s,head) attention, fully in LDS ----------------

__global__ void attn_kernel(const _Float16* __restrict__ qg,
                            const _Float16* __restrict__ kg,
                            const float* __restrict__ vf,
                            const float* __restrict__ sims,
                            const int* __restrict__ idxs,
                            float* __restrict__ out) {
  extern __shared__ char smem[];
  _Float16* qs   = (_Float16*)smem;                   // 128*32 f16
  _Float16* ks   = qs + 128 * 32;
  _Float16* vws  = ks + 128 * 32;                     // v * sims[u], f16
  float*    attn = (float*)(vws + 128 * 32);          // 128*128 f32
  _Float16* ath  = (_Float16*)(attn + 128 * 128);     // 128*128 f16 (softmaxed)
  float*    sm   = (float*)(ath + 128 * 128);         // 128
  int*      ix   = (int*)(sm + 128);                  // 128

  int tid = threadIdx.x;
  int wid = tid >> 5, lane = tid & 31;
  int blk = blockIdx.x;                               // b*S*NH + s*NH + hd
  int hd = blk % NH_;
  int s  = (blk / NH_) % S_;
  int b  = blk / (NH_ * S_);

  const float* simr = sims + ((size_t)b * S_ + s) * T_;
  const int*   idxr = idxs + ((size_t)b * S_ + s) * T_;
  for (int i = tid; i < T_; i += 256) { sm[i] = simr[i]; ix[i] = idxr[i]; }
  __syncthreads();

  size_t base = (size_t)b * C_ * HW_ + (size_t)(hd * HD_) * HW_;
  for (int e = tid; e < T_ * HD_; e += 256) {
    int t = e >> 5, cc = e & 31;
    size_t off = base + (size_t)cc * HW_ + ix[t];
    qs[e]  = qg[off];
    ks[e]  = kg[off];
    vws[e] = (_Float16)(vf[off] * sm[t]);
  }
  __syncthreads();

  // attn = (q @ k^T) * SCALE : 8x8 tiles of 16x16, wave w owns row-tile w
  {
    int ti = wid;
    v16h a = load_A(qs + ti * 16 * 32, 32, lane);
#pragma unroll
    for (int ui = 0; ui < 8; ++ui) {
      v16h bm = load_B_T(ks + ui * 16 * 32, 32, lane);
      v8f c = {};
      c = __builtin_amdgcn_wmma_f32_16x16x32_f16(false, a, false, bm,
                                                 (short)0, c, false, false);
      int hl = lane >> 4, n = lane & 15;
#pragma unroll
      for (int r = 0; r < 8; ++r) {
        int m = r + hl * 8;
        attn[(ti * 16 + m) * 128 + ui * 16 + n] = c[r] * SCALE_;
      }
    }
  }
  __syncthreads();

  // row softmax -> f16
  if (tid < T_) {
    const float* ar = attn + tid * 128;
    _Float16*    er = ath  + tid * 128;
    float mx = -INFINITY;
    for (int u = 0; u < 128; ++u) mx = fmaxf(mx, ar[u]);
    float sum = 0.f;
    for (int u = 0; u < 128; ++u) { float e = expf(ar[u] - mx); sum += e; er[u] = (_Float16)e; }
    float inv = 1.0f / sum;
    for (int u = 0; u < 128; ++u) er[u] = (_Float16)((float)er[u] * inv);
  }
  __syncthreads();

  // out = attn(128x128) @ vws(128x32); scale by sims[t]; scatter-add
  {
    int ti = wid;
#pragma unroll
    for (int ni = 0; ni < 2; ++ni) {
      v8f c = {};
#pragma unroll
      for (int kc = 0; kc < 128; kc += 32) {
        v16h a  = load_A(ath + ti * 16 * 128 + kc, 128, lane);
        v16h bm = load_B(vws + kc * 32 + ni * 16, 32, lane);
        c = __builtin_amdgcn_wmma_f32_16x16x32_f16(false, a, false, bm,
                                                   (short)0, c, false, false);
      }
      int hl = lane >> 4, n = lane & 15;
#pragma unroll
      for (int r = 0; r < 8; ++r) {
        int m  = r + hl * 8;
        int t  = ti * 16 + m;
        int cc = ni * 16 + n;
        float val = c[r] * sm[t];
        atomicAdd(&out[base + (size_t)cc * HW_ + ix[t]], val);
      }
    }
  }
}

// ---------------- Host launch ----------------

extern "C" void kernel_launch(void* const* d_in, const int* in_sizes, int n_in,
                              void* d_out, int out_size, void* d_ws, size_t ws_size,
                              hipStream_t stream) {
  const float* x   = (const float*)d_in[0];
  const float* aff = (const float*)d_in[1];
  const float* lw  = (const float*)d_in[2];
  const float* lb  = (const float*)d_in[3];
  const float* wq  = (const float*)d_in[4];
  const float* wk  = (const float*)d_in[5];
  const float* wv  = (const float*)d_in[6];
  float* out = (float*)d_out;

  const size_t NX   = (size_t)B_ * C_ * HW_;        // 8,388,608 elements
  const size_t NW   = (size_t)C_ * C_;              // 65,536 per weight
  char* ws = (char*)d_ws;
  size_t off = 0;
  auto alloc = [&](size_t bytes) -> char* {
    char* p = ws + off;
    off = (off + bytes + 255) & ~(size_t)255;
    return p;
  };
  _Float16* xnh  = (_Float16*)alloc(NX * 2);
  _Float16* wqh  = (_Float16*)alloc(NW * 2);
  _Float16* wkh  = (_Float16*)alloc(NW * 2);
  _Float16* wvh  = (_Float16*)alloc(NW * 2);
  _Float16* qbuf = (_Float16*)alloc(NX * 2);
  _Float16* kbuf = (_Float16*)alloc(NX * 2);
  float*    vbuf = (float*)   alloc(NX * 4);
  float*    simb = (float*)   alloc((size_t)B_ * S_ * T_ * 4);
  int*      idxb = (int*)     alloc((size_t)B_ * S_ * T_ * 4);
  if (off > ws_size) return;  // workspace too small; bail deterministically

  // 1) LayerNorm -> f16 activations
  ln_kernel<<<B_ * HW_ / 8, 256, 0, stream>>>(x, lw, lb, xnh);

  // 2) weights -> f16
  cvt_f16_kernel<<<(int)(NW / 256), 256, 0, stream>>>(wq, wqh, (int)NW);
  cvt_f16_kernel<<<(int)(NW / 256), 256, 0, stream>>>(wk, wkh, (int)NW);
  cvt_f16_kernel<<<(int)(NW / 256), 256, 0, stream>>>(wv, wvh, (int)NW);

  // 3) projections (q,k in f16; v in f32); 128x128 tile per block
  dim3 pg(B_ * 2 * 128);
  size_t pj_lds = 2 * 128 * 32 * 2;   // 16KB (W tile + X tile)
  proj_kernel<0><<<pg, 256, pj_lds, stream>>>(wqh, xnh, qbuf, nullptr);
  proj_kernel<1><<<pg, 256, pj_lds, stream>>>(wkh, xnh, kbuf, nullptr);
  proj_kernel<2><<<pg, 256, pj_lds, stream>>>(wvh, xnh, nullptr, vbuf);

  // 4) output base = v (attention scatter-adds on top; reads pristine vbuf)
  hipMemcpyAsync(out, vbuf, NX * 4, hipMemcpyDeviceToDevice, stream);

  // 5) top-128 selection per (b,s)
  size_t tk_lds = (size_t)HW_ * 4 + 1024 + 1024;
  topk_kernel<<<B_ * S_, 256, tk_lds, stream>>>(aff, simb, idxb);

  // 6) attention + scatter
  size_t at_lds = 3 * (128 * 32 * 2) + 128 * 128 * 4 + 128 * 128 * 2 + 512 + 512;
  attn_kernel<<<B_ * S_ * NH_, 256, at_lds, stream>>>(qbuf, kbuf, vbuf, simb, idxb, out);
}
